// Sinkhorn_81046032875887
// MI455X (gfx1250) — compile-verified, matrix operand order checked by reading, not run on hardware
//
#include <hip/hip_runtime.h>

#define N      8192
#define NITER  50
#define INVEPS 10.0f   // 1/EPSILON

typedef __attribute__((ext_vector_type(16))) _Float16 v16h;
typedef __attribute__((ext_vector_type(8)))  _Float16 v8h;
typedef __attribute__((ext_vector_type(8)))  float    v8f;

// ---------------------------------------------------------------------------
// Kernel 1: Kh[i,j] = (fp16) exp(-C[i,j]/eps).  128 MB fp16 -> L2 resident.
// ---------------------------------------------------------------------------
__global__ void __launch_bounds__(256) build_kh(const float* __restrict__ C,
                                                _Float16* __restrict__ Kh) {
    const long i = ((long)blockIdx.x * 256 + threadIdx.x) * 8;
    float4 c0 = *(const float4*)(C + i);
    float4 c1 = *(const float4*)(C + i + 4);
    v8h h;
    h[0] = (_Float16)__expf(-INVEPS * c0.x);
    h[1] = (_Float16)__expf(-INVEPS * c0.y);
    h[2] = (_Float16)__expf(-INVEPS * c0.z);
    h[3] = (_Float16)__expf(-INVEPS * c0.w);
    h[4] = (_Float16)__expf(-INVEPS * c1.x);
    h[5] = (_Float16)__expf(-INVEPS * c1.y);
    h[6] = (_Float16)__expf(-INVEPS * c1.z);
    h[7] = (_Float16)__expf(-INVEPS * c1.w);
    *(v8h*)(Kh + i) = h;
}

__global__ void __launch_bounds__(256) init_v(float* __restrict__ v,
                                              _Float16* __restrict__ vh) {
    int i = blockIdx.x * 256 + threadIdx.x;
    if (i < N) {
        v[i]  = 1.0f / (float)N;
        vh[i] = (_Float16)(1.0f / (float)N);
    }
}

// ---------------------------------------------------------------------------
// Kernel 2: u = 1 / (K @ v).
// Block = 8 waves, owns 16 rows; wave w reduces k-slice [w*1024, w*1024+1024).
// A = Kh rows (two v8h loads/lane per 16-bit A layout),
// B = vh chunk broadcast into all 16 columns (two v8h loads, no converts)
// -> D rows replicated across N.
// ---------------------------------------------------------------------------
__global__ void __launch_bounds__(256) sinkhorn_rows(const _Float16* __restrict__ Kh,
                                                     const _Float16* __restrict__ vh,
                                                     float* __restrict__ u,
                                                     _Float16* __restrict__ uh) {
    const int lane = threadIdx.x & 31;
    const int wave = threadIdx.x >> 5;
    const int hh   = lane >> 4;        // half-wave: 0 or 1
    const int m    = lane & 15;        // row within tile
    const int koff = hh * 8;           // 16-bit layout: lanes>=16 hold K+8

    const long row = (long)blockIdx.x * 16 + m;
    const long k0  = (long)wave * (N / 8);

    v8f c = {};
    const _Float16* ap = Kh + row * N + k0 + koff;
    const _Float16* vp = vh + k0 + koff;

    for (int kc = 0; kc < (N / 8) / 32; ++kc) {
        v16h a, b;
        v8h alo = *(const v8h*)(ap);
        v8h ahi = *(const v8h*)(ap + 16);
        v8h blo = *(const v8h*)(vp);
        v8h bhi = *(const v8h*)(vp + 16);
#pragma unroll
        for (int j = 0; j < 8; ++j) {
            a[j] = alo[j]; a[8 + j] = ahi[j];
            b[j] = blo[j]; b[8 + j] = bhi[j];   // B[k][n] = v[k] (all columns)
        }
        c = __builtin_amdgcn_wmma_f32_16x16x32_f16(false, a, false, b,
                                                   (short)0, c, false, false);
        ap += 32;
        vp += 32;
    }

    // D layout: VGPR r, lanes 0-15 -> M=r ; lanes 16-31 -> M=r+8 (N replicated)
    __shared__ float red[8][16];
    if (m == 0) {                      // lanes 0 and 16 write their 8 rows
#pragma unroll
        for (int r = 0; r < 8; ++r) red[wave][r + hh * 8] = c[r];
    }
    __syncthreads();
    if (threadIdx.x < 16) {
        float s = 0.0f;
#pragma unroll
        for (int w = 0; w < 8; ++w) s += red[w][threadIdx.x];
        float r = 1.0f / s;
        long  o = (long)blockIdx.x * 16 + threadIdx.x;
        u[o]  = r;
        uh[o] = (_Float16)r;
    }
}

// ---------------------------------------------------------------------------
// Kernel 3: v = 1 / (K^T @ u).
// Block owns 16 columns; wave w reduces row-slice [w*1024, ...).
// A = uh chunk broadcast into all rows (two v8h loads),
// B = 32x16 row-major block of Kh (per-lane column gather; row-strided d16
// loads hit L2 since Kh is resident; strides fit 24-bit IOFFSET so each load
// is base+imm with no per-load VALU) -> D columns replicated across M.
// ---------------------------------------------------------------------------
__global__ void __launch_bounds__(256) sinkhorn_cols(const _Float16* __restrict__ Kh,
                                                     const _Float16* __restrict__ uh,
                                                     float* __restrict__ v,
                                                     _Float16* __restrict__ vh) {
    const int lane = threadIdx.x & 31;
    const int wave = threadIdx.x >> 5;
    const int hh   = lane >> 4;
    const int n    = lane & 15;        // column within tile
    const int koff = hh * 8;           // 16-bit layout: lanes>=16 hold K+8

    const long col = (long)blockIdx.x * 16 + n;
    const long m0  = (long)wave * (N / 8);

    v8f c = {};
    const _Float16* bp = Kh + (m0 + koff) * N + col;
    const _Float16* up = uh + m0 + koff;

    for (int kc = 0; kc < (N / 8) / 32; ++kc) {
        v16h a, b;
        v8h alo = *(const v8h*)(up);
        v8h ahi = *(const v8h*)(up + 16);
#pragma unroll
        for (int j = 0; j < 8; ++j) {
            b[j]     = bp[(long)j * N];          // B[k][n] = Kh[m0+k][col]
            b[8 + j] = bp[(long)(16 + j) * N];
            a[j]     = alo[j];                   // A[m][k] = u[m0+k] (all rows)
            a[8 + j] = ahi[j];
        }
        c = __builtin_amdgcn_wmma_f32_16x16x32_f16(false, a, false, b,
                                                   (short)0, c, false, false);
        bp += (long)32 * N;
        up += 32;
    }

    // D rows replicated; lane n (0-15), VGPR 0 holds the column partial sum.
    __shared__ float red[8][16];
    if (hh == 0) red[wave][n] = c[0];
    __syncthreads();
    if (threadIdx.x < 16) {
        float s = 0.0f;
#pragma unroll
        for (int w = 0; w < 8; ++w) s += red[w][threadIdx.x];
        float r = 1.0f / s;
        long  o = (long)blockIdx.x * 16 + threadIdx.x;
        v[o]  = r;
        vh[o] = (_Float16)r;
    }
}

// ---------------------------------------------------------------------------
// Kernel 4: out[i,j] = u[i] * exp(-C[i,j]/eps) * v[j]   (full fp32 from C)
// Overwrites d_out (which held fp16 K during iterations; K is dead here).
// ---------------------------------------------------------------------------
__global__ void __launch_bounds__(256) final_plan(const float* __restrict__ C,
                                                  const float* __restrict__ u,
                                                  const float* __restrict__ v,
                                                  float* __restrict__ out) {
    const long e   = ((long)blockIdx.x * 256 + threadIdx.x) * 4;
    const long row = e >> 13;          // / 8192
    const long col = e & (N - 1);
    const float uu = u[row];
    float4 cc = *(const float4*)(C + e);
    float4 vv = *(const float4*)(v + col);
    float4 o;
    o.x = uu * __expf(-INVEPS * cc.x) * vv.x;
    o.y = uu * __expf(-INVEPS * cc.y) * vv.y;
    o.z = uu * __expf(-INVEPS * cc.z) * vv.z;
    o.w = uu * __expf(-INVEPS * cc.w) * vv.w;
    *(float4*)(out + e) = o;
}

// ---------------------------------------------------------------------------
extern "C" void kernel_launch(void* const* d_in, const int* in_sizes, int n_in,
                              void* d_out, int out_size, void* d_ws, size_t ws_size,
                              hipStream_t stream) {
    const float* C = (const float*)d_in[0];
    float* out = (float*)d_out;

    // fp16 K (128 MB) lives in d_out's storage until final_plan overwrites it.
    _Float16* Kh = (_Float16*)d_out;

    // Vectors in workspace: fp32 u,v for the final rescale; fp16 shadows for
    // the WMMA operand loads (96 KB total, 16B-aligned sub-buffers).
    float*     u  = (float*)d_ws;
    float*     v  = u + N;
    _Float16*  uh = (_Float16*)(v + N);
    _Float16*  vh = uh + N;

    build_kh<<<(long)N * N / (256 * 8), 256, 0, stream>>>(C, Kh);
    init_v<<<N / 256, 256, 0, stream>>>(v, vh);

    for (int it = 0; it < NITER; ++it) {
        sinkhorn_rows<<<N / 16, 256, 0, stream>>>(Kh, vh, u, uh);  // u = 1/(K v)
        sinkhorn_cols<<<N / 16, 256, 0, stream>>>(Kh, uh, v, vh);  // v = 1/(K^T u)
    }

    final_plan<<<(long)N * N / (256 * 4), 256, 0, stream>>>(C, u, v, out);
}
